// MultiKAN_695784702447
// MI455X (gfx1250) — matrix operand order repaired and checked by reading, not compiled, not run on hardware
//
#include <hip/hip_runtime.h>
#include <cstdint>
#include <cstddef>

// ---------------------------------------------------------------------------
// MultiKAN on MI455X (gfx1250, wave32, WMMA bf16 16x16x32, f32 accumulate)
//
// Each KAN layer is one GEMM:
//   out[b,o] = sum_i sb[i,o]*silu(x[b,i]) + sum_{i,k} B[b,i,k]*(sp[i,o]*coef[i,o,k])
//            = Aact(1024 x 4608) @ Wt(512 x 4608)^T
// Split-K by 4 for WGP utilization (256 blocks), async global->LDS staging
// (ASYNCcnt path) overlapped with WMMA, deterministic split-K reduction.
// ---------------------------------------------------------------------------

typedef __attribute__((ext_vector_type(16))) __bf16 v16bf;
typedef __attribute__((ext_vector_type(8)))  float  v8f;
typedef __attribute__((ext_vector_type(4)))  int    v4i_;

#define IN_DIM   512
#define OUT_DIM  512
#define BATCH_N  1024
#define NBASIS   8                  // GRID_SIZE + K
#define NGRID    12                 // (GRID_SIZE+1) + 2K knots
#define KPACK    9                  // 1 silu + 8 basis values
#define KDIM     (IN_DIM * KPACK)   // 4608

#define BM 128
#define BN 64
#define BK 32
#define LDSPAD 40                   // 32 u16 + 8 u16 pad: 80B rows, 16B-aligned chunks
#define STEPS (KDIM / BK)           // 144
#define SPLITK 4
#define KSPS  (STEPS / SPLITK)      // 36 K-steps per split

#if __has_builtin(__builtin_amdgcn_global_load_async_to_lds_b128)
#define USE_ASYNC 1
#else
#define USE_ASYNC 0
#endif

static __device__ __forceinline__ void wait_async0() {
#if __has_builtin(__builtin_amdgcn_s_wait_asynccnt)
    __builtin_amdgcn_s_wait_asynccnt(0);
#else
    asm volatile("s_wait_asynccnt 0" ::: "memory");
#endif
}

#if USE_ASYNC
static __device__ __forceinline__ void async_b128(const void* g, void* l) {
    // signature (from hipcc diagnostic): (int4 AS1* gsrc, int4 AS3* lds, imm offset, imm cpol)
    __builtin_amdgcn_global_load_async_to_lds_b128(
        (__attribute__((address_space(1))) v4i_*)g,
        (__attribute__((address_space(3))) v4i_*)l,
        0, 0);
}
#endif

static __device__ __forceinline__ unsigned short f2bf(float f) {
    // round-to-nearest-even f32 -> bf16 (raw bits)
    unsigned u = __float_as_uint(f);
    unsigned r = u + 0x7FFFu + ((u >> 16) & 1u);
    return (unsigned short)(r >> 16);
}

// ---------------------------------------------------------------------------
// Activation prep: per (b,i) compute silu(x) and the 8 cubic B-spline basis
// values (Cox-de Boor, 12 knots), emit 9 bf16 values into Aact[b][i*9+c].
// ---------------------------------------------------------------------------
__global__ void kan_prep_act(const float* __restrict__ x,
                             const float* __restrict__ grid,
                             unsigned short* __restrict__ Aact) {
    const int idx = blockIdx.x * blockDim.x + threadIdx.x;   // b*512 + i
    const int i = idx & (IN_DIM - 1);
    const float xv = x[idx];

    const float* __restrict__ gr = grid + i * NGRID;
    float g[NGRID];
#pragma unroll
    for (int j = 0; j < NGRID; ++j) g[j] = gr[j];

    float Bv[NGRID - 1];
#pragma unroll
    for (int j = 0; j < NGRID - 1; ++j)
        Bv[j] = (xv >= g[j] && xv < g[j + 1]) ? 1.0f : 0.0f;
#pragma unroll
    for (int p = 1; p <= 3; ++p)
#pragma unroll
        for (int j = 0; j < NGRID - 2; ++j)
            if (j < NGRID - 1 - p) {
                const float lw = (xv - g[j]) / (g[j + p] - g[j]);
                const float rw = (g[j + p + 1] - xv) / (g[j + p + 1] - g[j + 1]);
                Bv[j] = lw * Bv[j] + rw * Bv[j + 1];
            }

    const float silu = xv / (1.0f + __expf(-xv));

    unsigned short* __restrict__ dst = Aact + (size_t)(idx >> 9) * KDIM + i * KPACK;
    dst[0] = f2bf(silu);
#pragma unroll
    for (int k = 0; k < NBASIS; ++k) dst[1 + k] = f2bf(Bv[k]);
}

// ---------------------------------------------------------------------------
// Weight packing: Wt row-major [OUT_DIM][KDIM] bf16:
//   Wt[o][i*9+0] = sb[i,o],  Wt[o][i*9+1+k] = sp[i,o]*coef[i,o,k]
// ---------------------------------------------------------------------------
__global__ void kan_pack_w(const float* __restrict__ coef,
                           const float* __restrict__ sb,
                           const float* __restrict__ sp,
                           unsigned short* __restrict__ Wt) {
    const int idx = blockIdx.x * blockDim.x + threadIdx.x;   // i*512 + o
    const int o = idx & (OUT_DIM - 1);
    const int i = idx >> 9;
    const float sbv = sb[(size_t)i * OUT_DIM + o];
    const float spv = sp[(size_t)i * OUT_DIM + o];
    const float* __restrict__ c = coef + ((size_t)i * OUT_DIM + o) * NBASIS;
    unsigned short* __restrict__ dst = Wt + (size_t)o * KDIM + i * KPACK;
    dst[0] = f2bf(sbv);
#pragma unroll
    for (int k = 0; k < NBASIS; ++k) dst[1 + k] = f2bf(spv * c[k]);
}

// ---------------------------------------------------------------------------
// Split-K GEMM: part[z] = Aact(128 x 1152) @ Wt^T per block tile.
// 256 threads = 8 waves (4x2), each wave owns 2x2 16x16 fragments.
// Async global->LDS double-buffered staging overlapped with WMMA.
// ---------------------------------------------------------------------------
__global__ __launch_bounds__(256) void kan_gemm(const unsigned short* __restrict__ A,
                                                const unsigned short* __restrict__ Wt,
                                                float* __restrict__ part) {
    __shared__ __align__(16) unsigned short As[2][BM][LDSPAD];
    __shared__ __align__(16) unsigned short Bs[2][BN][LDSPAD];

    const int tid  = threadIdx.x;
    const int lane = tid & 31;
    const int waveM = (tid >> 5) & 3;    // 4 waves along M
    const int waveN = (tid >> 5) >> 2;   // 2 waves along N

    const int mBase = blockIdx.y * BM;
    const int nBase = blockIdx.x * BN;
    const int kStart = blockIdx.z * KSPS;
    const int kEnd   = kStart + KSPS;

    // staging: A tile = 512 x 16B chunks (2/thread), B tile = 256 (1/thread)
    const int aRow0 = tid >> 2;
    const int aRow1 = (tid + 256) >> 2;
    const int aCol  = (tid & 3) * 8;     // u16 offset of this 16B chunk in the row
    const unsigned short* __restrict__ gA0 = A  + (size_t)(mBase + aRow0) * KDIM + aCol;
    const unsigned short* __restrict__ gA1 = A  + (size_t)(mBase + aRow1) * KDIM + aCol;
    const unsigned short* __restrict__ gB  = Wt + (size_t)(nBase + aRow0) * KDIM + aCol;

    // fragment read mapping (CDNA5 16-bit A 16x32 / B 32x16 VGPR layouts)
    const int mRow = waveM * 32 + (lane & 15);
    const int aK0  = (lane >> 4) * 8;    // lanes 16-31 start at K=8
    const int nCol = waveN * 32 + (lane & 15);
    const int bK0  = (lane >> 4) * 8;    // lanes 16-31 start at K=8

    v8f acc[2][2] = {};

    // stage tile kt into LDS buffer b
    auto stage = [&](int kt, int b) {
        const size_t off = (size_t)kt * BK;
#if USE_ASYNC
        async_b128(gA0 + off, &As[b][aRow0][aCol]);
        async_b128(gA1 + off, &As[b][aRow1][aCol]);
        async_b128(gB  + off, &Bs[b][aRow0][aCol]);
#else
        const uint4 ra0 = *(const uint4*)(gA0 + off);
        const uint4 ra1 = *(const uint4*)(gA1 + off);
        const uint4 rb  = *(const uint4*)(gB  + off);
        *(uint4*)&As[b][aRow0][aCol] = ra0;
        *(uint4*)&As[b][aRow1][aCol] = ra1;
        *(uint4*)&Bs[b][aRow0][aCol] = rb;
#endif
    };

    stage(kStart, 0);
#if USE_ASYNC
    wait_async0();
#endif
    __syncthreads();

    int buf = 0;
    for (int kt = kStart; kt < kEnd; ++kt) {
        const bool hasNext = (kt + 1) < kEnd;
        if (hasNext) stage(kt + 1, buf ^ 1);        // DMA next tile (fire & forget)
        if (kt + 4 < kEnd) {                        // warm L2 further ahead
            __builtin_prefetch(gA0 + (size_t)(kt + 4) * BK, 0, 1);
            __builtin_prefetch(gB  + (size_t)(kt + 4) * BK, 0, 1);
        }

        union Frag { uint4 q[2]; v16bf v; };
        Frag af[2], bfr[2];
#pragma unroll
        for (int mf = 0; mf < 2; ++mf) {
            const unsigned short* p = &As[buf][mRow + mf * 16][0];
            af[mf].q[0] = *(const uint4*)(p + aK0);        // K k0..k0+7
            af[mf].q[1] = *(const uint4*)(p + aK0 + 16);   // K k0+16..k0+23
        }
#pragma unroll
        for (int nf = 0; nf < 2; ++nf) {
            const unsigned short* p = &Bs[buf][nCol + nf * 16][0];
            bfr[nf].q[0] = *(const uint4*)(p + bK0);       // K k0..k0+7
            bfr[nf].q[1] = *(const uint4*)(p + bK0 + 16);  // K k0+16..k0+23
        }
#pragma unroll
        for (int mf = 0; mf < 2; ++mf)
#pragma unroll
            for (int nf = 0; nf < 2; ++nf)
                acc[mf][nf] = __builtin_amdgcn_wmma_f32_16x16x32_bf16(
                    false, af[mf].v, false, bfr[nf].v,
                    (short)0, acc[mf][nf], false, false);

#if USE_ASYNC
        if (hasNext) wait_async0();                 // DMA done before barrier
#endif
        __syncthreads();
        buf ^= 1;
    }

    // epilogue: C/D layout -> VGPR r holds M=r (lanes 0-15) / M=r+8 (lanes 16-31)
    float* __restrict__ outp = part + (size_t)blockIdx.z * (BATCH_N * OUT_DIM);
#pragma unroll
    for (int mf = 0; mf < 2; ++mf)
#pragma unroll
        for (int nf = 0; nf < 2; ++nf) {
            const int col   = nBase + waveN * 32 + nf * 16 + (lane & 15);
            const int rbase = mBase + waveM * 32 + mf * 16 + ((lane >> 4) << 3);
#pragma unroll
            for (int r = 0; r < 8; ++r)
                outp[(size_t)(rbase + r) * OUT_DIM + col] = acc[mf][nf][r];
        }
}

// ---------------------------------------------------------------------------
// Deterministic split-K reduction: out = sum_z part[z], vectorized float4.
// ---------------------------------------------------------------------------
__global__ void kan_reduce(const float* __restrict__ part, float* __restrict__ out) {
    const int idx = blockIdx.x * blockDim.x + threadIdx.x;   // over M*N/4
    const float4* __restrict__ p = (const float4*)part;
    const int stride = (BATCH_N * OUT_DIM) / 4;
    float4 a = p[idx];
#pragma unroll
    for (int s = 1; s < SPLITK; ++s) {
        const float4 b = p[idx + s * stride];
        a.x += b.x; a.y += b.y; a.z += b.z; a.w += b.w;
    }
    ((float4*)out)[idx] = a;
}

// ---------------------------------------------------------------------------
// Host side
// ---------------------------------------------------------------------------
extern "C" void kernel_launch(void* const* d_in, const int* in_sizes, int n_in,
                              void* d_out, int out_size, void* d_ws, size_t ws_size,
                              hipStream_t stream) {
    (void)in_sizes; (void)n_in; (void)out_size; (void)ws_size;

    // workspace layout (16B aligned):
    //   Aact : 1024*4608 bf16 = 9,437,184 B
    //   Wt   :  512*4608 bf16 = 4,718,592 B
    //   xbuf : 1024*512  f32  = 2,097,152 B
    //   part : 4*1024*512 f32 = 8,388,608 B      (total ~24.6 MB)
    unsigned short* Aact = (unsigned short*)d_ws;
    unsigned short* Wt   = (unsigned short*)((char*)d_ws + 9437184u);
    float*          xbuf = (float*)((char*)d_ws + 9437184u + 4718592u);
    float*          part = (float*)((char*)d_ws + 9437184u + 4718592u + 2097152u);

    const float* x = (const float*)d_in[0];
    for (int l = 0; l < 3; ++l) {
        const float* grid = (const float*)d_in[1 + 4 * l + 0];
        const float* coef = (const float*)d_in[1 + 4 * l + 1];
        const float* sb   = (const float*)d_in[1 + 4 * l + 2];
        const float* sp   = (const float*)d_in[1 + 4 * l + 3];
        float* y = (l == 2) ? (float*)d_out : xbuf;

        kan_prep_act<<<(BATCH_N * IN_DIM) / 256, 256, 0, stream>>>(x, grid, Aact);
        kan_pack_w<<<(IN_DIM * OUT_DIM) / 256, 256, 0, stream>>>(coef, sb, sp, Wt);
        kan_gemm<<<dim3(OUT_DIM / BN, BATCH_N / BM, SPLITK), 256, 0, stream>>>(Aact, Wt, part);
        kan_reduce<<<(BATCH_N * OUT_DIM / 4) / 256, 256, 0, stream>>>(part, y);

        x = y;   // layer l output feeds layer l+1 (Aact consumed before overwrite)
    }
}
